// AttnEncoder_61125974556731
// MI455X (gfx1250) — matrix operand order repaired and checked by reading, not compiled, not run on hardware
//
#include <hip/hip_runtime.h>
#include <hip/hip_bf16.h>
#include <stdint.h>

// Problem constants (match reference).
#define BB 64
#define TT 512
#define DD 256
#define G4D 1024   // 4*D
#define VV 32000

typedef __attribute__((ext_vector_type(16))) __bf16 v16bf;
typedef __attribute__((ext_vector_type(8)))  __bf16 v8bf;
typedef __attribute__((ext_vector_type(8)))  float  v8f;

// Build a 16x32 bf16 A-fragment (contiguous K order, same as B):
// lane holds row M = lane%16; lanes 0-15 carry K = kb..kb+15,
// lanes 16-31 carry K = kb+16..kb+31. Caller passes the two
// contiguous 8-element runs for this lane.
static __device__ inline v16bf make_a_frag(const __bf16* lo, const __bf16* hi) {
    v8bf a = *reinterpret_cast<const v8bf*>(lo);
    v8bf b = *reinterpret_cast<const v8bf*>(hi);
    v16bf r;
#pragma unroll
    for (int i = 0; i < 8; ++i) { r[i] = a[i]; r[8 + i] = b[i]; }
    return r;
}

static __device__ inline float sigf(float x) {
    return 1.0f / (1.0f + __expf(-x));
}

// ---------------------------------------------------------------------------
// Kernel 1: convert weights to bf16, fold b_ih + b_hh into one bias vector.
// ---------------------------------------------------------------------------
__global__ void prep_kernel(const float* __restrict__ W_ih,
                            const float* __restrict__ W_hh,
                            const float* __restrict__ b_ih,
                            const float* __restrict__ b_hh,
                            __bf16* __restrict__ wih_b,
                            __bf16* __restrict__ whh_b,
                            float* __restrict__ bias) {
    int i = blockIdx.x * blockDim.x + threadIdx.x;
    if (i < G4D * DD) {
        wih_b[i] = (__bf16)W_ih[i];
        whh_b[i] = (__bf16)W_hh[i];
    }
    if (i < G4D) bias[i] = b_ih[i] + b_hh[i];
}

// ---------------------------------------------------------------------------
// Kernel 2: Gx[b*T + t][g] = emb[words[b][t]] . W_ih[g] + bias[g]
// One 16-row M-tile per blockIdx.x, 128 gate-columns per blockIdx.y,
// 8 waves x one 16x16 WMMA tile each, K-loop of 8 (256 = 8*32).
// ---------------------------------------------------------------------------
__global__ __launch_bounds__(256) void embed_gemm_kernel(
    const int*   __restrict__ words,
    const float* __restrict__ emb,
    const __bf16* __restrict__ wih_b,
    const float* __restrict__ bias,
    float* __restrict__ Gx) {
    __shared__ __align__(16) __bf16 Xs[16 * DD];   // 8 KB staged activations

    const int mtile = blockIdx.x;   // 0..2047  (rows of B*T)
    const int cgrp  = blockIdx.y;   // 0..7     (128 gate cols each)
    const int tid   = threadIdx.x;

    // Stage 16 gathered embedding rows as bf16 into LDS.
    {
        const int row = tid >> 4;          // 0..15
        const int k0  = (tid & 15) << 4;   // 0..240 step 16
        const int n   = mtile * 16 + row;
        const int w   = words[n];
        const float4* src = reinterpret_cast<const float4*>(emb + (size_t)w * DD + k0);
        __bf16* dst = &Xs[row * DD + k0];
#pragma unroll
        for (int q = 0; q < 4; ++q) {
            float4 v = src[q];
            dst[q * 4 + 0] = (__bf16)v.x;
            dst[q * 4 + 1] = (__bf16)v.y;
            dst[q * 4 + 2] = (__bf16)v.z;
            dst[q * 4 + 3] = (__bf16)v.w;
        }
    }
    __syncthreads();

    const int wave = tid >> 5;
    const int lane = tid & 31;
    const int hlf  = lane >> 4;                          // lane half selects K sub-block
    const int col  = cgrp * 128 + wave * 16 + (lane & 15);  // global gate column
    const int arow = lane & 15;

    // Seed accumulator with bias (same column for all 8 M rows in this lane).
    float bb = bias[col];
    v8f acc;
#pragma unroll
    for (int r = 0; r < 8; ++r) acc[r] = bb;

#pragma unroll
    for (int kc = 0; kc < 8; ++kc) {
        const int kb = kc * 32;
        v16bf a = make_a_frag(&Xs[arow * DD + kb + hlf * 16],
                              &Xs[arow * DD + kb + hlf * 16 + 8]);
        // B fragment: column = col, 16 contiguous K values starting at kb + hlf*16.
        v16bf b = *reinterpret_cast<const v16bf*>(wih_b + (size_t)col * DD + kb + hlf * 16);
        acc = __builtin_amdgcn_wmma_f32_16x16x32_bf16(
            false, a, false, b, (short)0, acc, false, false);
    }

    // D layout: row M = r + 8*hlf, col = col.
#pragma unroll
    for (int r = 0; r < 8; ++r) {
        const int m = r + 8 * hlf;
        Gx[(size_t)(mtile * 16 + m) * G4D + col] = acc[r];
    }
}

// ---------------------------------------------------------------------------
// Kernel 3: the two LSTM scans. Batches are independent, so the grid is
// (4 batch-groups of 16) x (2 directions) = 8 workgroups (8 WGPs busy on the
// latency-bound recurrence). Per block: 16 waves, one (16-batch x 16-hidden)
// task per wave, all 4 gate accumulator tiles in registers so the gate
// nonlinearity never crosses waves. h lives in LDS (16x256 bf16 = 8 KB),
// c lives in registers, W_hh (bf16, 512 KB) streams from L2 each step.
// Reverse direction gathers Gx at clip(len-1-t, 0).
// ---------------------------------------------------------------------------
__global__ __launch_bounds__(512, 1) void lstm_kernel(
    const int*   __restrict__ lengths,
    const __bf16* __restrict__ whh_b,
    const float* __restrict__ Gx,
    float* __restrict__ out_vec,    // 256 floats: hr_final[-1]
    float* __restrict__ out_all) {  // T*B*512: [h_fwd | h_rev]
    __shared__ __align__(16) __bf16 Hs[16 * DD];   // 8 KB hidden state (this group)

    const int grp  = blockIdx.x;        // 0..3 batch group (rows grp*16..+15)
    const int dir  = blockIdx.y;        // 0 = forward, 1 = reverse
    const int tid  = threadIdx.x;
    const int lane = tid & 31;
    const int wave = tid >> 5;          // 0..15 = j tile
    const int hlf  = lane >> 4;
    const int l16  = lane & 15;
    const int m0   = grp * 16;
    const int j0   = wave * 16;
    const int colj = j0 + l16;          // hidden column owned by this lane

    // Zero initial hidden state.
    {
        uint32_t* hz = reinterpret_cast<uint32_t*>(Hs);
#pragma unroll
        for (int i = tid; i < 16 * DD / 2; i += 512) hz[i] = 0u;
    }

    int lenr[8];
    v8f cst;
#pragma unroll
    for (int r = 0; r < 8; ++r) {
        lenr[r] = lengths[m0 + r + 8 * hlf];
        cst[r]  = 0.0f;
    }
    __syncthreads();

    for (int t = 0; t < TT; ++t) {
        // ---- Phase 1: seed accumulators from Gx (bias already folded in).
        v8f accI, accF, accG, accO;
#pragma unroll
        for (int r = 0; r < 8; ++r) {
            const int b = m0 + r + 8 * hlf;
            int st = t;
            if (dir) { st = lenr[r] - 1 - t; if (st < 0) st = 0; }
            const size_t rowb = ((size_t)b * TT + st) * G4D;
            accI[r] = Gx[rowb + 0 * DD + colj];
            accF[r] = Gx[rowb + 1 * DD + colj];
            accG[r] = Gx[rowb + 2 * DD + colj];
            accO[r] = Gx[rowb + 3 * DD + colj];
        }

        // h(t-1) @ W_hh^T: one shared A fragment per K chunk feeds all 4 gates.
#pragma unroll
        for (int kc = 0; kc < 8; ++kc) {
            const int kb = kc * 32;
            v16bf a = make_a_frag(&Hs[l16 * DD + kb + hlf * 16],
                                  &Hs[l16 * DD + kb + hlf * 16 + 8]);
            const size_t kof = (size_t)kb + hlf * 16;
            v16bf bi = *reinterpret_cast<const v16bf*>(
                whh_b + (size_t)(0 * DD + colj) * DD + kof);
            v16bf bf_ = *reinterpret_cast<const v16bf*>(
                whh_b + (size_t)(1 * DD + colj) * DD + kof);
            v16bf bg = *reinterpret_cast<const v16bf*>(
                whh_b + (size_t)(2 * DD + colj) * DD + kof);
            v16bf bo = *reinterpret_cast<const v16bf*>(
                whh_b + (size_t)(3 * DD + colj) * DD + kof);
            accI = __builtin_amdgcn_wmma_f32_16x16x32_bf16(
                false, a, false, bi, (short)0, accI, false, false);
            accF = __builtin_amdgcn_wmma_f32_16x16x32_bf16(
                false, a, false, bf_, (short)0, accF, false, false);
            accG = __builtin_amdgcn_wmma_f32_16x16x32_bf16(
                false, a, false, bg, (short)0, accG, false, false);
            accO = __builtin_amdgcn_wmma_f32_16x16x32_bf16(
                false, a, false, bo, (short)0, accO, false, false);
        }

        // Prefetch next step's Gx seed rows (global_prefetch_b8 path) —
        // the only HBM-streaming traffic inside the recurrence.
        if (t + 1 < TT) {
            const int b0 = m0 + 8 * hlf;            // r == 0 row of this half
            int st1 = t + 1;
            if (dir) { st1 = lenr[0] - 2 - t; if (st1 < 0) st1 = 0; }
            const size_t rowb1 = ((size_t)b0 * TT + st1) * G4D;
#pragma unroll
            for (int g = 0; g < 4; ++g) {
                __builtin_prefetch(&Gx[rowb1 + g * DD + colj], 0, 1);
            }
        }
        __syncthreads();   // all reads of Hs(t-1) done

        // ---- Phase 2: gate nonlinearity, state update, masked output.
#pragma unroll
        for (int r = 0; r < 8; ++r) {
            const int lb = r + 8 * hlf;     // local batch row 0..15
            const int b  = m0 + lb;
            const bool m = (t < lenr[r]);
            const float gi = sigf(accI[r]);
            const float gf = sigf(accF[r]);
            const float gg = tanhf(accG[r]);
            const float go = sigf(accO[r]);
            const float c2 = gf * cst[r] + gi * gg;
            const float h2 = go * tanhf(c2);
            if (m) {
                cst[r] = c2;
                Hs[lb * DD + colj] = (__bf16)h2;
            }
            out_all[((size_t)t * BB + b) * (2 * DD) + dir * DD + colj] =
                m ? h2 : 0.0f;
        }
        __syncthreads();   // Hs(t) visible before next step's reads
    }

    // hr_final[-1] = final reverse hidden state of batch 63 (grp 3, local 15).
    if (dir == 1 && grp == 3 && tid < DD) {
        out_vec[tid] = (float)Hs[15 * DD + tid];
    }
}

// ---------------------------------------------------------------------------
// Host launcher.
// ---------------------------------------------------------------------------
extern "C" void kernel_launch(void* const* d_in, const int* in_sizes, int n_in,
                              void* d_out, int out_size, void* d_ws, size_t ws_size,
                              hipStream_t stream) {
    const int*   words   = (const int*)d_in[0];
    const int*   lengths = (const int*)d_in[1];
    const float* emb     = (const float*)d_in[2];
    const float* W_ih    = (const float*)d_in[3];
    const float* W_hh    = (const float*)d_in[4];
    const float* b_ih    = (const float*)d_in[5];
    const float* b_hh    = (const float*)d_in[6];
    float* out = (float*)d_out;

    // Workspace layout (≈135 MB):
    //   [0, 512K)        W_ih bf16
    //   [512K, 1M)       W_hh bf16
    //   [1M, 1M+4K)      bias f32
    //   [1M+4K, +128M)   Gx f32  (B*T x 4D, bias folded in)
    char* ws = (char*)d_ws;
    __bf16* wih_b = (__bf16*)(ws);
    __bf16* whh_b = (__bf16*)(ws + 524288);
    float*  bias  = (float*)(ws + 1048576);
    float*  Gx    = (float*)(ws + 1052672);

    prep_kernel<<<(G4D * DD + 255) / 256, 256, 0, stream>>>(
        W_ih, W_hh, b_ih, b_hh, wih_b, whh_b, bias);

    embed_gemm_kernel<<<dim3(BB * TT / 16, G4D / 128), 256, 0, stream>>>(
        words, emb, wih_b, bias, Gx);

    lstm_kernel<<<dim3(4, 2), 512, 0, stream>>>(
        lengths, whh_b, Gx, out, out + DD);
}
